// DiscriminativeLossVectorized_32976758898732
// MI455X (gfx1250) — matrix operand order repaired and checked by reading, not compiled, not run on hardware
//
#include <hip/hip_runtime.h>

typedef _Float16 v16h __attribute__((ext_vector_type(16)));
typedef float    v8f  __attribute__((ext_vector_type(8)));

#define NUM_B   4
#define NUM_E   16
#define NUM_C   32
#define N_PIX   (64 * 128 * 128)   // 1,048,576 pixels per batch
#define TILE_PIX 32
#define TILES_PER_BLOCK 64         // 2048 pixels per (1-wave) block
#define CHUNKS   (N_PIX / (TILE_PIX * TILES_PER_BLOCK))  // 512

#define EPS_F    1e-12f
#define DVAR_F   0.5f
#define DDIST_F  1.5f

// ---------------- workspace layout (floats) ----------------
#define WS_SUMS    0      // [B][32][16] (2048)
#define WS_COUNTS  2048   // [B][32]     (128)
#define WS_HINGED  2176   // [B][32]     (128)
#define WS_CENTERS 2304   // [B][32][16] (2048)
#define WS_DIST    4352   // [B]
#define WS_REG     4356   // [B]
#define WS_NINST   4360   // [B]
#define WS_ZERO_N  2304   // sums+counts+hinged must start at zero

__global__ void zero_ws_kernel(float* w, int n) {
    for (int i = threadIdx.x + blockIdx.x * blockDim.x; i < n;
         i += blockDim.x * gridDim.x)
        w[i] = 0.0f;
}

// ---------------------------------------------------------------------------
// Pass 1: segment sums + counts via one-hot WMMA.
//   sums[32x16]  += onehot(ids)[32xK] @ emb[Kx16]
//   counts[32]    = row-sums of onehot = onehot @ ones[Kx16]  (any column)
// Block = 1 wave (32 threads).  Embedding tile staged by async LDS DMA
// (global_load_async_to_lds_b128, ASYNCcnt), 4 instructions per 2KB tile.
// ---------------------------------------------------------------------------
__global__ void __launch_bounds__(32)
seg_sum_kernel(const float* __restrict__ emb, const int* __restrict__ mask,
               float* __restrict__ sums, float* __restrict__ counts) {
    const int b    = blockIdx.y;
    const int lane = threadIdx.x;                 // 0..31
    const size_t pix_base = (size_t)blockIdx.x * (TILE_PIX * TILES_PER_BLOCK);

    const float* embB  = emb  + (size_t)b * NUM_E * N_PIX;
    const int*   maskB = mask + (size_t)b * N_PIX;

    __shared__ float tilef[NUM_E][TILE_PIX];      // 16 x 32 f32 (2KB), DMA target
    __shared__ int   ids[TILE_PIX];

    // low 32 bits of a generic LDS pointer == byte offset within the
    // workgroup's LDS allocation, which is what VDST of the async load expects
    const unsigned lds_tile = (unsigned)(unsigned long long)(uintptr_t)&tilef[0][0];

    v8f acc0  = {};   // ids  0..15  x dims 0..15
    v8f acc1  = {};   // ids 16..31  x dims 0..15
    v8f accC0 = {};   // counts ids  0..15 (replicated over columns)
    v8f accC1 = {};   // counts ids 16..31

    v16h onesv;
#pragma unroll
    for (int j = 0; j < 16; ++j) onesv[j] = (_Float16)1.0f;

    const int m     = lane & 15;         // my id-row / dim-col
    const int rsub  = lane >> 3;         // 0..3: sub-row for DMA
    const int c4    = (lane & 7) * 4;    // float column of my 16B chunk
    const int khalf = (lane < 16) ? 0 : 1;

    for (int t = 0; t < TILES_PER_BLOCK; ++t) {
        const size_t p0 = pix_base + (size_t)t * TILE_PIX;

        ids[lane] = maskB[p0 + lane];

        // ---- async DMA: 16 rows x 128B contiguous; 4 rows per instruction ----
#pragma unroll
        for (int g = 0; g < 4; ++g) {
            const int n = g * 4 + rsub;
            const float* gp = embB + (size_t)n * N_PIX + p0 + c4;
            const unsigned lo = lds_tile + (unsigned)((n * TILE_PIX + c4) * 4);
            asm volatile("global_load_async_to_lds_b128 %0, %1, off"
                         :: "v"(lo), "v"(gp) : "memory");
        }
        asm volatile("s_wait_asynccnt 0x0" ::: "memory");
        __syncthreads();   // dscnt fence for ids[] (1-wave block: barrier ~free)

        // ---- build fragments: A and B both use contiguous k order (k = 16h + s) ----
        // A (16x32, MxK): lanes 0-15 hold K=0..15;         lanes 16-31 K=16..31
        // B (32x16, KxN): lanes 0-15 hold K=0..15;         lanes 16-31 K=16..31
        v16h a0, a1, bf;
        {
            const float4* rp = (const float4*)&tilef[m][khalf * 16];
            const float4 r0 = rp[0], r1 = rp[1], r2 = rp[2], r3 = rp[3];
            bf[0]  = (_Float16)r0.x;  bf[1]  = (_Float16)r0.y;
            bf[2]  = (_Float16)r0.z;  bf[3]  = (_Float16)r0.w;
            bf[4]  = (_Float16)r1.x;  bf[5]  = (_Float16)r1.y;
            bf[6]  = (_Float16)r1.z;  bf[7]  = (_Float16)r1.w;
            bf[8]  = (_Float16)r2.x;  bf[9]  = (_Float16)r2.y;
            bf[10] = (_Float16)r2.z;  bf[11] = (_Float16)r2.w;
            bf[12] = (_Float16)r3.x;  bf[13] = (_Float16)r3.y;
            bf[14] = (_Float16)r3.z;  bf[15] = (_Float16)r3.w;
        }
#pragma unroll
        for (int j = 0; j < 16; ++j) {
            const int kA  = khalf * 16 + j;
            const int idk = ids[kA];
            a0[j] = (idk == m)      ? (_Float16)1.0f : (_Float16)0.0f;
            a1[j] = (idk == m + 16) ? (_Float16)1.0f : (_Float16)0.0f;
        }

        acc0  = __builtin_amdgcn_wmma_f32_16x16x32_f16(
                    false, a0, false, bf,    (short)0, acc0,  false, false);
        acc1  = __builtin_amdgcn_wmma_f32_16x16x32_f16(
                    false, a1, false, bf,    (short)0, acc1,  false, false);
        accC0 = __builtin_amdgcn_wmma_f32_16x16x32_f16(
                    false, a0, false, onesv, (short)0, accC0, false, false);
        accC1 = __builtin_amdgcn_wmma_f32_16x16x32_f16(
                    false, a1, false, onesv, (short)0, accC1, false, false);
        __syncthreads();   // tile reads complete before next DMA overwrites
    }

    // ---- D layout: VGPR r, lane L -> M = r + (L<16?0:8), N = L&15 ----
    const int mrow = (lane < 16) ? 0 : 8;
    const int n    = lane & 15;
#pragma unroll
    for (int r = 0; r < 8; ++r) {
        atomicAdd(&sums[((size_t)b * NUM_C + (r + mrow))      * NUM_E + n], acc0[r]);
        atomicAdd(&sums[((size_t)b * NUM_C + (r + mrow + 16)) * NUM_E + n], acc1[r]);
    }
    if (n == 0) {   // count matrix columns are identical; lanes 0 and 16 cover M 0..15
#pragma unroll
        for (int r = 0; r < 8; ++r) {
            atomicAdd(&counts[b * NUM_C + (r + mrow)],      accC0[r]);
            atomicAdd(&counts[b * NUM_C + (r + mrow + 16)], accC1[r]);
        }
    }
}

// ---------------------------------------------------------------------------
// Pass 2 (tiny): centers, pairwise dist loss, reg loss, num_inst. 1 block/batch.
// ---------------------------------------------------------------------------
__global__ void centers_kernel(const float* __restrict__ sums,
                               const float* __restrict__ counts,
                               float* __restrict__ centers,
                               float* __restrict__ dist_out,
                               float* __restrict__ reg_out,
                               float* __restrict__ ninst_out) {
    const int b = blockIdx.x;
    const int t = threadIdx.x;   // 64 threads
    __shared__ float cent[NUM_C][NUM_E];
    __shared__ float cnts[NUM_C];
    __shared__ float s_ld, s_np, s_lr, s_ni;

    if (t == 0) { s_ld = 0.f; s_np = 0.f; s_lr = 0.f; s_ni = 0.f; }
    if (t < NUM_C) cnts[t] = counts[b * NUM_C + t];
    __syncthreads();

    for (int i = t; i < NUM_C * NUM_E; i += 64) {
        const int c = i >> 4, e = i & 15;
        const float v = sums[((size_t)b * NUM_C + c) * NUM_E + e] / fmaxf(cnts[c], 1.0f);
        cent[c][e] = v;
        centers[((size_t)b * NUM_C + c) * NUM_E + e] = v;
    }
    __syncthreads();

    float ld = 0.f, np = 0.f;
    for (int pidx = t; pidx < 31 * 31; pidx += 64) {
        const int i = pidx / 31 + 1, j = pidx % 31 + 1;
        if (i < j && cnts[i] > 0.f && cnts[j] > 0.f) {
            float s = 0.f;
#pragma unroll
            for (int e = 0; e < NUM_E; ++e) {
                const float d = cent[i][e] - cent[j][e];
                s += d * d;
            }
            const float cd = sqrtf(s + EPS_F);
            const float h  = fmaxf(2.0f * DDIST_F - cd, 0.f);
            ld += h * h;
            np += 1.f;
        }
    }

    float lr = 0.f, ni = 0.f;
    if (t >= 1 && t < NUM_C && cnts[t] > 0.f) {
        float s = 0.f;
#pragma unroll
        for (int e = 0; e < NUM_E; ++e) s += cent[t][e] * cent[t][e];
        lr = sqrtf(s + EPS_F);
        ni = 1.f;
    }

    atomicAdd(&s_ld, ld); atomicAdd(&s_np, np);
    atomicAdd(&s_lr, lr); atomicAdd(&s_ni, ni);
    __syncthreads();
    if (t == 0) {
        dist_out[b]  = s_ld / fmaxf(s_np, 1.f);
        reg_out[b]   = s_lr / fmaxf(s_ni, 1.f);
        ninst_out[b] = s_ni;
    }
}

// ---------------------------------------------------------------------------
// Pass 3: hinged variance term.  4 pixels/lane via b128 loads (512B/wave/instr),
// centers cached in LDS, per-block LDS hinged accumulation.
// ---------------------------------------------------------------------------
__global__ void __launch_bounds__(256)
var_kernel(const float* __restrict__ emb, const int* __restrict__ mask,
           const float* __restrict__ centers, float* __restrict__ hinged) {
    const int b = blockIdx.y;
    const int t = threadIdx.x;  // 256
    __shared__ float cent[NUM_C * NUM_E];
    __shared__ float hacc[NUM_C];

    for (int i = t; i < NUM_C * NUM_E; i += 256)
        cent[i] = centers[(size_t)b * NUM_C * NUM_E + i];
    if (t < NUM_C) hacc[t] = 0.f;
    __syncthreads();

    const float* embB  = emb  + (size_t)b * NUM_E * N_PIX;
    const int*   maskB = mask + (size_t)b * N_PIX;
    const size_t stride4 = (size_t)gridDim.x * 256 * 4;

    for (size_t p = ((size_t)blockIdx.x * 256 + t) * 4; p < N_PIX; p += stride4) {
        __builtin_prefetch(&maskB[p + stride4], 0, 1);   // global_prefetch_b8
        const int4 idv = *(const int4*)(maskB + p);
        float sx = 0.f, sy = 0.f, sz = 0.f, sw = 0.f;
#pragma unroll
        for (int n = 0; n < NUM_E; ++n) {
            const float4 ev = *(const float4*)(embB + (size_t)n * N_PIX + p);
            const float dx = ev.x - cent[idv.x * NUM_E + n]; sx += dx * dx;
            const float dy = ev.y - cent[idv.y * NUM_E + n]; sy += dy * dy;
            const float dz = ev.z - cent[idv.z * NUM_E + n]; sz += dz * dz;
            const float dw = ev.w - cent[idv.w * NUM_E + n]; sw += dw * dw;
        }
        const float hx = fmaxf(sqrtf(sx + EPS_F) - DVAR_F, 0.f);
        const float hy = fmaxf(sqrtf(sy + EPS_F) - DVAR_F, 0.f);
        const float hz = fmaxf(sqrtf(sz + EPS_F) - DVAR_F, 0.f);
        const float hw = fmaxf(sqrtf(sw + EPS_F) - DVAR_F, 0.f);
        atomicAdd(&hacc[idv.x], hx * hx);
        atomicAdd(&hacc[idv.y], hy * hy);
        atomicAdd(&hacc[idv.z], hz * hz);
        atomicAdd(&hacc[idv.w], hw * hw);
    }
    __syncthreads();
    if (t < NUM_C) atomicAdd(&hinged[b * NUM_C + t], hacc[t]);
}

// ---------------------------------------------------------------------------
// Pass 4 (tiny): combine everything into the 4 output scalars.
// ---------------------------------------------------------------------------
__global__ void finalize_kernel(const float* __restrict__ counts,
                                const float* __restrict__ hinged,
                                const float* __restrict__ dist,
                                const float* __restrict__ regv,
                                const float* __restrict__ ninst,
                                float* __restrict__ out) {
    const int t = threadIdx.x;  // 128 = 4 batches x 32 ids
    __shared__ float lv[NUM_B];
    if (t < NUM_B) lv[t] = 0.f;
    __syncthreads();

    const int b = t >> 5, c = t & 31;
    float v = 0.f;
    if (c >= 1) {
        const float cntv = counts[b * NUM_C + c];
        if (cntv > 0.f) v = hinged[b * NUM_C + c] / fmaxf(cntv, 1.f);
    }
    atomicAdd(&lv[b], v);
    __syncthreads();

    if (t == 0) {
        float LV = 0.f, LD = 0.f, LR = 0.f;
        for (int bb = 0; bb < NUM_B; ++bb) {
            LV += lv[bb] / fmaxf(ninst[bb], 1.f);
            LD += dist[bb];
            LR += regv[bb];
        }
        LV *= (1.0f / NUM_B); LD *= (1.0f / NUM_B); LR *= (1.0f / NUM_B);
        out[0] = 1.0f * LV + 1.0f * LD + 0.001f * LR;  // ALPHA, BETA, GAMMA
        out[1] = LV;
        out[2] = LD;
        out[3] = LR;
    }
}

extern "C" void kernel_launch(void* const* d_in, const int* in_sizes, int n_in,
                              void* d_out, int out_size, void* d_ws, size_t ws_size,
                              hipStream_t stream) {
    (void)in_sizes; (void)n_in; (void)out_size; (void)ws_size;
    const float* emb  = (const float*)d_in[0];
    const int*   mask = (const int*)d_in[1];
    float* out = (float*)d_out;
    float* w   = (float*)d_ws;

    float* sums    = w + WS_SUMS;
    float* counts  = w + WS_COUNTS;
    float* hinged  = w + WS_HINGED;
    float* centers = w + WS_CENTERS;
    float* dist    = w + WS_DIST;
    float* regv    = w + WS_REG;
    float* ninst   = w + WS_NINST;

    zero_ws_kernel<<<8, 256, 0, stream>>>(w, WS_ZERO_N);

    dim3 g1(CHUNKS, NUM_B);
    seg_sum_kernel<<<g1, 32, 0, stream>>>(emb, mask, sums, counts);

    centers_kernel<<<NUM_B, 64, 0, stream>>>(sums, counts, centers, dist, regv, ninst);

    dim3 g3(512, NUM_B);
    var_kernel<<<g3, 256, 0, stream>>>(emb, mask, centers, hinged);

    finalize_kernel<<<1, 128, 0, stream>>>(counts, hinged, dist, regv, ninst, out);
}
